// NodeClassifierGNN_gat_19078244729018
// MI455X (gfx1250) — matrix operand order repaired and checked by reading, not compiled, not run on hardware
//
#include <hip/hip_runtime.h>
#include <hip/hip_bf16.h>
#include <stdint.h>

// ---------------------------------------------------------------------------
// GAT (2 layers, H=4 heads x C=32) + linear head for MI455X / gfx1250.
// GEMMs: v_wmma_f32_16x16x32_f16 (f16 in, f32 accum). Edge softmax/aggregation:
// f32 atomics (memory-bound part, ~1.8 GB traffic -> ~80us at 23.3 TB/s).
// ---------------------------------------------------------------------------

typedef __attribute__((ext_vector_type(16))) _Float16 v16h;
typedef __attribute__((ext_vector_type(8)))  _Float16 v8h;
typedef __attribute__((ext_vector_type(8)))  float    v8f;

#define HCF 128   // feature width (IN == H*C == 128)
#define NHD 4     // heads

__device__ __forceinline__ float leaky02(float x) { return x > 0.f ? x : 0.2f * x; }

// order-preserving float<->uint encoding for atomicMax on signed floats
__device__ __forceinline__ unsigned f2ord(float f) {
  unsigned u = __float_as_uint(f);
  return (u & 0x80000000u) ? ~u : (u | 0x80000000u);
}
__device__ __forceinline__ float ord2f(unsigned u) {
  return __uint_as_float((u & 0x80000000u) ? (u ^ 0x80000000u) : ~u);
}

// -------------------------- conversion kernels -----------------------------

__global__ void f32_to_f16_kernel(const float* __restrict__ in,
                                  _Float16* __restrict__ out, int n) {
  int i = blockIdx.x * blockDim.x + threadIdx.x;
  if (i < n) out[i] = (_Float16)in[i];
}

// Wt[n][k] = (f16) W[k][n]  (128x128), so B-fragment loads are k-contiguous
__global__ void wtrans_f16_kernel(const float* __restrict__ W,
                                  _Float16* __restrict__ Wt) {
  int t = blockIdx.x * blockDim.x + threadIdx.x;
  if (t < HCF * HCF) {
    int n = t >> 7, k = t & 127;
    Wt[t] = (_Float16)W[k * HCF + n];
  }
}

__global__ void fill_u32_kernel(unsigned* __restrict__ p, unsigned v, int n) {
  int i = blockIdx.x * blockDim.x + threadIdx.x;
  if (i < n) p[i] = v;
}

// ------------------------------- WMMA GEMM ---------------------------------
// C[M x 128] = A[M x 128] (f16) * W (via Wt[n][k] f16), f32 accumulate.
// Block = 256 threads = 8 waves; block -> 16 rows, wave w -> cols [16w,16w+16).
__global__ __launch_bounds__(256) void gat_gemm_wmma(
    const _Float16* __restrict__ A,   // [M x 128] row-major f16
    const _Float16* __restrict__ Bt,  // [128 x 128] f16, Bt[n*128+k] = W[k][n]
    float* __restrict__ C,            // [M x 128] f32
    int M) {
  const int lane = threadIdx.x & 31;
  const int wave = threadIdx.x >> 5;
  const int row0 = blockIdx.x << 4;
  const int col0 = wave << 4;
  const int m16  = lane & 15;
  const int hi   = lane >> 4;

  int r = row0 + m16; if (r >= M) r = M - 1;   // clamp (EXEC must stay all-ones)
  // A 16-bit 16x32 fragment: lane m holds K = hi*8 + [0..7] and K = hi*8+16 + [0..7]
  const _Float16* ap = A  + (size_t)r * HCF + (hi << 3);
  // B 32x16 fragment: lane n holds K = hi*8 + [0..7] and K = hi*8+16 + [0..7]
  const _Float16* bp = Bt + (size_t)(col0 + m16) * HCF + (hi << 3);

  v8f acc = {};
#pragma unroll
  for (int ks = 0; ks < 128; ks += 32) {
    v8h a_lo = *(const v8h*)(ap + ks);
    v8h a_hi = *(const v8h*)(ap + ks + 16);
    v8h b_lo = *(const v8h*)(bp + ks);
    v8h b_hi = *(const v8h*)(bp + ks + 16);
    v16h a = __builtin_shufflevector(a_lo, a_hi, 0,1,2,3,4,5,6,7,8,9,10,11,12,13,14,15);
    v16h b = __builtin_shufflevector(b_lo, b_hi, 0,1,2,3,4,5,6,7,8,9,10,11,12,13,14,15);
    acc = __builtin_amdgcn_wmma_f32_16x16x32_f16(false, a, false, b,
                                                 (short)0, acc, false, false);
  }

  // C/D layout: lane (n = lane&15, hi), VGPR v -> row M = hi*8 + v
  const int mb = hi << 3;
  float* cp = C + (size_t)(row0 + mb) * HCF + col0 + m16;
  if (row0 + 16 <= M) {
    // fast path: whole 16-row tile in range -> straight-line stores
#pragma unroll
    for (int v = 0; v < 8; ++v) cp[(size_t)v * HCF] = acc[v];
  } else {
#pragma unroll
    for (int v = 0; v < 8; ++v) {
      if (row0 + mb + v < M) cp[(size_t)v * HCF] = acc[v];
    }
  }
}

// ------------------------- per-node attention scores -----------------------
// s_src[n][h] = sum_c h[n][h*32+c] * a_src[h*32+c]  (one wave per node)
__global__ __launch_bounds__(256) void scores_kernel(
    const float* __restrict__ h, const float* __restrict__ a_src,
    const float* __restrict__ a_dst, float* __restrict__ ssrc,
    float* __restrict__ sdst, int N) {
  const int lane = threadIdx.x & 31;
  const int n = blockIdx.x * (blockDim.x >> 5) + (threadIdx.x >> 5);
  if (n >= N) return;
  float4 hv = *(const float4*)(h + (size_t)n * HCF + lane * 4);
  float4 as = *(const float4*)(a_src + lane * 4);
  float4 ad = *(const float4*)(a_dst + lane * 4);
  float ps = hv.x * as.x + hv.y * as.y + hv.z * as.z + hv.w * as.w;
  float pd = hv.x * ad.x + hv.y * ad.y + hv.z * ad.z + hv.w * ad.w;
  // reduce within each 8-lane head group
  ps += __shfl_xor(ps, 4, 32); pd += __shfl_xor(pd, 4, 32);
  ps += __shfl_xor(ps, 2, 32); pd += __shfl_xor(pd, 2, 32);
  ps += __shfl_xor(ps, 1, 32); pd += __shfl_xor(pd, 1, 32);
  if ((lane & 7) == 0) {
    int hh = lane >> 3;
    ssrc[(size_t)n * NHD + hh] = ps;
    sdst[(size_t)n * NHD + hh] = pd;
  }
}

// ----------------------------- edge passes ---------------------------------

__global__ void edge_max_kernel(const int* __restrict__ src,
                                const int* __restrict__ dst,
                                const float* __restrict__ ssrc,
                                const float* __restrict__ sdst,
                                unsigned* __restrict__ mord, int E, int N) {
  int e = blockIdx.x * blockDim.x + threadIdx.x;
  if (e >= E + N) return;
  int s, d;
  if (e < E) { s = src[e]; d = dst[e]; } else { s = e - E; d = s; }
  float4 a = *(const float4*)(ssrc + 4 * (size_t)s);
  float4 b = *(const float4*)(sdst + 4 * (size_t)d);
  unsigned* mp = mord + 4 * (size_t)d;
  atomicMax(mp + 0, f2ord(leaky02(a.x + b.x)));
  atomicMax(mp + 1, f2ord(leaky02(a.y + b.y)));
  atomicMax(mp + 2, f2ord(leaky02(a.z + b.z)));
  atomicMax(mp + 3, f2ord(leaky02(a.w + b.w)));
}

__global__ void mfin_kernel(const unsigned* __restrict__ mord,
                            float* __restrict__ mf, int n) {
  int i = blockIdx.x * blockDim.x + threadIdx.x;
  if (i >= n) return;
  float v = ord2f(mord[i]);
  if (!(v >= -3.0e38f && v <= 3.0e38f)) v = 0.f;  // where(isfinite(m), m, 0)
  mf[i] = v;
}

__global__ void edge_sum_kernel(const int* __restrict__ src,
                                const int* __restrict__ dst,
                                const float* __restrict__ ssrc,
                                const float* __restrict__ sdst,
                                const float* __restrict__ mf,
                                float* __restrict__ z, int E, int N) {
  int e = blockIdx.x * blockDim.x + threadIdx.x;
  if (e >= E + N) return;
  int s, d;
  if (e < E) { s = src[e]; d = dst[e]; } else { s = e - E; d = s; }
  float4 a = *(const float4*)(ssrc + 4 * (size_t)s);
  float4 b = *(const float4*)(sdst + 4 * (size_t)d);
  float4 m = *(const float4*)(mf + 4 * (size_t)d);
  float* zp = z + 4 * (size_t)d;
  atomicAdd(zp + 0, __expf(leaky02(a.x + b.x) - m.x));
  atomicAdd(zp + 1, __expf(leaky02(a.y + b.y) - m.y));
  atomicAdd(zp + 2, __expf(leaky02(a.z + b.z) - m.z));
  atomicAdd(zp + 3, __expf(leaky02(a.w + b.w) - m.w));
}

// one wave per edge; lane l handles channels [4l, 4l+4) (head = l>>3)
__global__ __launch_bounds__(256) void edge_agg_kernel(
    const int* __restrict__ src, const int* __restrict__ dst,
    const float* __restrict__ ssrc, const float* __restrict__ sdst,
    const float* __restrict__ mf, const float* __restrict__ z,
    const float* __restrict__ hmat, float* __restrict__ acc, int E, int N) {
  const int lane = threadIdx.x & 31;
  const int e = blockIdx.x * (blockDim.x >> 5) + (threadIdx.x >> 5);
  if (e >= E + N) return;
  int s, d;
  if (e < E) { s = src[e]; d = dst[e]; } else { s = e - E; d = s; }

  const float* hp = hmat + (size_t)s * HCF + lane * 4;
  __builtin_prefetch(hp, 0, 0);  // -> global_prefetch_b8, hide gather latency

  float4 a  = *(const float4*)(ssrc + 4 * (size_t)s);
  float4 b  = *(const float4*)(sdst + 4 * (size_t)d);
  float4 m  = *(const float4*)(mf + 4 * (size_t)d);
  float4 zz = *(const float4*)(z + 4 * (size_t)d);
  float al0 = __expf(leaky02(a.x + b.x) - m.x) / (zz.x + 1e-16f);
  float al1 = __expf(leaky02(a.y + b.y) - m.y) / (zz.y + 1e-16f);
  float al2 = __expf(leaky02(a.z + b.z) - m.z) / (zz.z + 1e-16f);
  float al3 = __expf(leaky02(a.w + b.w) - m.w) / (zz.w + 1e-16f);

  const int hs = lane >> 3;
  float alo   = (hs & 1) ? al1 : al0;
  float ahi   = (hs & 1) ? al3 : al2;
  float alpha = (hs & 2) ? ahi : alo;

  float4 hv = *(const float4*)hp;
  float* ap = acc + (size_t)d * HCF + lane * 4;
  atomicAdd(ap + 0, hv.x * alpha);
  atomicAdd(ap + 1, hv.y * alpha);
  atomicAdd(ap + 2, hv.z * alpha);
  atomicAdd(ap + 3, hv.w * alpha);
}

// bias + ReLU; write next-layer activations in both f32 and f16
__global__ void epilogue_kernel(const float* __restrict__ acc,
                                const float* __restrict__ bias,
                                float* __restrict__ actF,
                                _Float16* __restrict__ actH, int total) {
  int i = blockIdx.x * blockDim.x + threadIdx.x;
  if (i >= total) return;
  float v = acc[i] + bias[i & 127];
  v = v > 0.f ? v : 0.f;
  actF[i] = v;
  actH[i] = (_Float16)v;
}

// out[n] = dot(act[n, :128], lin_w) + lin_b   (one wave per node)
__global__ __launch_bounds__(256) void final_linear_kernel(
    const float* __restrict__ act, const float* __restrict__ w,
    const float* __restrict__ bptr, float* __restrict__ out, int N) {
  const int lane = threadIdx.x & 31;
  const int n = blockIdx.x * (blockDim.x >> 5) + (threadIdx.x >> 5);
  if (n >= N) return;
  float4 hv = *(const float4*)(act + (size_t)n * HCF + lane * 4);
  float4 wv = *(const float4*)(w + lane * 4);
  float s = hv.x * wv.x + hv.y * wv.y + hv.z * wv.z + hv.w * wv.w;
  s += __shfl_xor(s, 16, 32);
  s += __shfl_xor(s, 8, 32);
  s += __shfl_xor(s, 4, 32);
  s += __shfl_xor(s, 2, 32);
  s += __shfl_xor(s, 1, 32);
  if (lane == 0) out[n] = s + bptr[0];
}

// ------------------------------ host driver --------------------------------

extern "C" void kernel_launch(void* const* d_in, const int* in_sizes, int n_in,
                              void* d_out, int out_size, void* d_ws, size_t ws_size,
                              hipStream_t stream) {
  const float* x   = (const float*)d_in[0];
  const int*   src = (const int*)d_in[1];
  const int*   dst = (const int*)d_in[2];
  const float* W1  = (const float*)d_in[3];
  const float* as1 = (const float*)d_in[4];
  const float* ad1 = (const float*)d_in[5];
  const float* b1  = (const float*)d_in[6];
  const float* W2  = (const float*)d_in[7];
  const float* as2 = (const float*)d_in[8];
  const float* ad2 = (const float*)d_in[9];
  const float* b2  = (const float*)d_in[10];
  const float* lw  = (const float*)d_in[11];
  const float* lb  = (const float*)d_in[12];
  float* out = (float*)d_out;

  const int N  = in_sizes[0] / HCF;  // IN == 128
  const int E  = in_sizes[1];
  const int ET = E + N;              // edges + self loops
  (void)n_in; (void)out_size; (void)ws_size;

  // carve scratch from d_ws (all 256B-aligned)
  char* base = (char*)d_ws;
  size_t off = 0;
  auto take = [&](size_t bytes) -> char* {
    char* p = base + off;
    off = (off + bytes + 255) & ~(size_t)255;
    return p;
  };
  _Float16* actH = (_Float16*)take((size_t)N * HCF * sizeof(_Float16));
  _Float16* Wt1  = (_Float16*)take((size_t)HCF * HCF * sizeof(_Float16));
  _Float16* Wt2  = (_Float16*)take((size_t)HCF * HCF * sizeof(_Float16));
  float* hbuf = (float*)take((size_t)N * HCF * sizeof(float));
  float* acc  = (float*)take((size_t)N * HCF * sizeof(float));
  float* actF = (float*)take((size_t)N * HCF * sizeof(float));
  float* ssrc = (float*)take((size_t)N * NHD * sizeof(float));
  float* sdst = (float*)take((size_t)N * NHD * sizeof(float));
  unsigned* mord = (unsigned*)take((size_t)N * NHD * sizeof(unsigned));
  float* mf   = (float*)take((size_t)N * NHD * sizeof(float));
  float* zbuf = (float*)take((size_t)N * NHD * sizeof(float));

  const int TB = 256;
  const unsigned ORD_NEG_INF = 0x007FFFFFu;  // f2ord(-inf)

  // input conversions (once)
  f32_to_f16_kernel<<<(N * HCF + TB - 1) / TB, TB, 0, stream>>>(x, actH, N * HCF);
  wtrans_f16_kernel<<<(HCF * HCF + TB - 1) / TB, TB, 0, stream>>>(W1, Wt1);
  wtrans_f16_kernel<<<(HCF * HCF + TB - 1) / TB, TB, 0, stream>>>(W2, Wt2);

  for (int layer = 0; layer < 2; ++layer) {
    const _Float16* Wt = layer ? Wt2 : Wt1;
    const float* as = layer ? as2 : as1;
    const float* ad = layer ? ad2 : ad1;
    const float* bb = layer ? b2 : b1;

    gat_gemm_wmma<<<(N + 15) / 16, 256, 0, stream>>>(actH, Wt, hbuf, N);
    scores_kernel<<<(N + 7) / 8, 256, 0, stream>>>(hbuf, as, ad, ssrc, sdst, N);

    fill_u32_kernel<<<(N * NHD + TB - 1) / TB, TB, 0, stream>>>(mord, ORD_NEG_INF, N * NHD);
    fill_u32_kernel<<<(N * NHD + TB - 1) / TB, TB, 0, stream>>>((unsigned*)zbuf, 0u, N * NHD);
    fill_u32_kernel<<<(N * HCF + TB - 1) / TB, TB, 0, stream>>>((unsigned*)acc, 0u, N * HCF);

    edge_max_kernel<<<(ET + TB - 1) / TB, TB, 0, stream>>>(src, dst, ssrc, sdst, mord, E, N);
    mfin_kernel<<<(N * NHD + TB - 1) / TB, TB, 0, stream>>>(mord, mf, N * NHD);
    edge_sum_kernel<<<(ET + TB - 1) / TB, TB, 0, stream>>>(src, dst, ssrc, sdst, mf, zbuf, E, N);
    edge_agg_kernel<<<(ET + 7) / 8, 256, 0, stream>>>(src, dst, ssrc, sdst, mf, zbuf,
                                                      hbuf, acc, E, N);
    epilogue_kernel<<<(N * HCF + TB - 1) / TB, TB, 0, stream>>>(acc, bb, actF, actH, N * HCF);
  }

  final_linear_kernel<<<(N + 7) / 8, 256, 0, stream>>>(actF, lw, lb, out, N);
}
